// GNNPolicy_88459146428956
// MI455X (gfx1250) — matrix operand order repaired and checked by reading, not compiled, not run on hardware
//
#include <hip/hip_runtime.h>
#include <hip/hip_bf16.h>

typedef __attribute__((ext_vector_type(16))) _Float16 v16h;
typedef __attribute__((ext_vector_type(8)))  float    v8f;

#define HD 64      // hidden dim
#define FEAT 128   // input features
#define NGRAPH 64  // graphs in batch
#define TB 256

// ---------------- small utility kernels ----------------
__global__ void k_fill(float* __restrict__ p, float v, int n) {
  int i = blockIdx.x * blockDim.x + threadIdx.x;
  if (i < n) p[i] = v;
}

__global__ void k_deg(const int* __restrict__ dst, float* __restrict__ deg, int E) {
  int i = blockIdx.x * blockDim.x + threadIdx.x;
  if (i < E) atomicAdd(&deg[dst[i]], 1.0f);
}

__global__ void k_dinv(float* __restrict__ deg, int n) {
  int i = blockIdx.x * blockDim.x + threadIdx.x;
  if (i < n) {
    float d = deg[i];
    deg[i] = (d > 0.0f) ? rsqrtf(d) : 0.0f;
  }
}

__global__ void k_norm(const int* __restrict__ src, const int* __restrict__ dst,
                       const float* __restrict__ dinv, float* __restrict__ norm, int E) {
  int i = blockIdx.x * blockDim.x + threadIdx.x;
  if (i < E) norm[i] = dinv[src[i]] * dinv[dst[i]];
}

// ---------------- WMMA GEMM: out[N,64] = A[N,K] @ W[K,64] ----------------
// One wave computes a 16x64 output stripe (4 accumulators of 16x16).
// W is pre-packed in LDS in the per-lane B-fragment layout, so each
// fragment is ONE contiguous 32-byte LDS read per lane (ds_load_b128 x2).
template <int K>
__global__ void k_gemm_wmma(const float* __restrict__ A, const float* __restrict__ W,
                            float* __restrict__ out, int nrows) {
  constexpr int KT = K / 32;                    // k-tiles
  __shared__ v16h wlds[KT * 4 * 32];            // [ktile][ctile][lane] -> 16 halfs

  // cooperative pack: element i of lane ln, tile (kt,c) is W[(kt*32+boff+(i&7)+((i>>3)<<4))*HD + n]
  for (int t = threadIdx.x; t < KT * 4 * 32; t += blockDim.x) {
    int kt = t >> 7;
    int c  = (t >> 5) & 3;
    int ln = t & 31;
    int boff = (ln >= 16) ? 8 : 0;
    int n = c * 16 + (ln & 15);
    v16h tmp;
#pragma unroll
    for (int i = 0; i < 16; ++i)
      tmp[i] = (_Float16)W[(kt * 32 + boff + (i & 7) + ((i >> 3) << 4)) * HD + n];
    wlds[t] = tmp;
  }
  __syncthreads();

  int wave = blockIdx.x * (blockDim.x >> 5) + (threadIdx.x >> 5);
  int lane = threadIdx.x & 31;
  int row0 = wave * 16;
  if (row0 >= nrows) return;           // wave-uniform exit (EXEC stays full for WMMA)

  const bool hi = lane >= 16;
  const int hioff = hi ? 8 : 0;        // A K-offset / D row-offset for upper half-wave
  const int l16 = lane & 15;
  const int m = row0 + l16;            // A row this lane holds

  v8f acc[4] = {};

#pragma unroll
  for (int kt = 0; kt < KT; ++kt) {
    const int kk = kt * 32;
    // --- A fragment: two runs of 8 contiguous floats -> 4x global_load_b128 ---
    v16h a = {};
    if (m < nrows) {
      const float4* p0 = (const float4*)(A + (size_t)m * K + kk + hioff);
      const float4* p1 = (const float4*)(A + (size_t)m * K + kk + 16 + hioff);
      float4 u0 = p0[0], u1 = p0[1];
      float4 v0 = p1[0], v1 = p1[1];
      a[0]  = (_Float16)u0.x; a[1]  = (_Float16)u0.y; a[2]  = (_Float16)u0.z; a[3]  = (_Float16)u0.w;
      a[4]  = (_Float16)u1.x; a[5]  = (_Float16)u1.y; a[6]  = (_Float16)u1.z; a[7]  = (_Float16)u1.w;
      a[8]  = (_Float16)v0.x; a[9]  = (_Float16)v0.y; a[10] = (_Float16)v0.z; a[11] = (_Float16)v0.w;
      a[12] = (_Float16)v1.x; a[13] = (_Float16)v1.y; a[14] = (_Float16)v1.z; a[15] = (_Float16)v1.w;
    }
    // --- 4 column tiles: each B fragment is a single contiguous LDS vector ---
#pragma unroll
    for (int c = 0; c < 4; ++c) {
      v16h b = wlds[(kt * 4 + c) * 32 + lane];
      acc[c] = __builtin_amdgcn_wmma_f32_16x16x32_f16(
          false, a, false, b, (short)0, acc[c], false, false);
    }
  }

  // D layout: VGPR r -> row (r + hioff), lane%16 -> column
#pragma unroll
  for (int c = 0; c < 4; ++c) {
#pragma unroll
    for (int r = 0; r < 8; ++r) {
      int row = row0 + r + hioff;
      if (row < nrows)
        out[(size_t)row * HD + c * 16 + l16] = acc[c][r];
    }
  }
}

// ---------------- edge scatter: agg[dst] += xw[src] * norm ----------------
// one wave per edge; each lane handles 2 contiguous features (float2 gather)
__global__ void k_agg(const float* __restrict__ xw, const int* __restrict__ src,
                      const int* __restrict__ dst, const float* __restrict__ norm,
                      float* __restrict__ agg, int E) {
  long long gt = (long long)blockIdx.x * blockDim.x + threadIdx.x;
  int e = (int)(gt >> 5);
  int lane = threadIdx.x & 31;
  if (e < E) {
    int s = src[e], d = dst[e];
    float w = norm[e];
    float2 v = *((const float2*)(xw + ((size_t)s << 6)) + lane);
    float* q = agg + ((size_t)d << 6) + (lane << 1);
    atomicAdd(q,     v.x * w);
    atomicAdd(q + 1, v.y * w);
  }
}

// ---------------- self-loop + bias + relu (in place on agg) ----------------
__global__ void k_bias_relu(float* __restrict__ agg, const float* __restrict__ xw,
                            const float* __restrict__ dinv, const float* __restrict__ b,
                            int n) {
  int i = blockIdx.x * blockDim.x + threadIdx.x;
  if (i < n * HD) {
    int node = i >> 6;
    int f = i & 63;
    float di = dinv[node];
    float v = agg[i] + xw[i] * di * di + b[f];
    agg[i] = fmaxf(v, 0.0f);
  }
}

// ---------------- action logits: out[i] = h[i,:] . Wa + ba ----------------
__global__ void k_logits(const float* __restrict__ h, const float* __restrict__ Wa,
                         const float* __restrict__ ba, float* __restrict__ out, int n) {
  int i = blockIdx.x * blockDim.x + threadIdx.x;
  if (i >= n) return;
  const float4* hr = (const float4*)(h + (size_t)i * HD);
  const float4* wa = (const float4*)Wa;
  float s = 0.0f;
#pragma unroll
  for (int j = 0; j < HD / 4; ++j) {
    float4 a = hr[j], w = wa[j];
    s += a.x * w.x + a.y * w.y + a.z * w.z + a.w * w.w;
  }
  out[i] = s + ba[0];
}

// ---------------- per-graph mean pool + value head ----------------
__device__ __forceinline__ int lbound(const int* a, int n, int v) {
  int lo = 0, hi = n;
  while (lo < hi) {
    int mid = (lo + hi) >> 1;
    if (a[mid] < v) lo = mid + 1; else hi = mid;
  }
  return lo;
}

__global__ void k_value(const float* __restrict__ h, const int* __restrict__ batch,
                        const float* __restrict__ Wc, const float* __restrict__ bc,
                        float* __restrict__ out, int n) {
  __shared__ float part[TB];
  int g = blockIdx.x;
  int lo = lbound(batch, n, g);
  int hi = lbound(batch, n, g + 1);
  int t = threadIdx.x;
  int f = t & 63, chunk = t >> 6;
  float acc = 0.0f;
  for (int i = lo + chunk; i < hi; i += 4) acc += h[(size_t)i * HD + f];
  part[t] = acc;
  __syncthreads();
  if (t < HD) {
    float s = part[t] + part[t + 64] + part[t + 128] + part[t + 192];
    float cnt = (float)(hi - lo);
    s /= fmaxf(cnt, 1.0f);
    part[t] = s * Wc[t];
  }
  __syncthreads();
  if (t == 0) {
    float s = 0.0f;
    for (int i = 0; i < HD; ++i) s += part[i];
    out[g] = s + bc[0];
  }
}

// ---------------- launch ----------------
extern "C" void kernel_launch(void* const* d_in, const int* in_sizes, int n_in,
                              void* d_out, int out_size, void* d_ws, size_t ws_size,
                              hipStream_t stream) {
  const float* x    = (const float*)d_in[0];
  const int*   ei   = (const int*)d_in[1];
  const int*   batch= (const int*)d_in[2];
  const float* W1   = (const float*)d_in[3];
  const float* b1   = (const float*)d_in[4];
  const float* W2   = (const float*)d_in[5];
  const float* b2   = (const float*)d_in[6];
  const float* Wa   = (const float*)d_in[7];
  const float* ba   = (const float*)d_in[8];
  const float* Wc   = (const float*)d_in[9];
  const float* bc   = (const float*)d_in[10];

  const int N = in_sizes[2];
  const int E = in_sizes[1] / 2;
  const int* src = ei;
  const int* dst = ei + E;
  float* outv = (float*)d_out;

  float* ws   = (float*)d_ws;
  float* dinv = ws;                       // N   (deg, then deg^-1/2)
  float* norm = dinv + N;                 // E
  float* bufA = norm + E;                 // N*64
  float* bufB = bufA + (size_t)N * HD;    // N*64

  auto cdiv = [](long long a, long long b) { return (int)((a + b - 1) / b); };

  // --- normalization coefficients ---
  k_fill<<<cdiv(N, TB), TB, 0, stream>>>(dinv, 1.0f, N);       // self-loop degree
  k_deg <<<cdiv(E, TB), TB, 0, stream>>>(dst, dinv, E);
  k_dinv<<<cdiv(N, TB), TB, 0, stream>>>(dinv, N);
  k_norm<<<cdiv(E, TB), TB, 0, stream>>>(src, dst, dinv, norm, E);

  const int tiles = cdiv(N, 16);
  const int gemmBlocks = cdiv(tiles, TB / 32);
  const int nhBlocks = cdiv((long long)N * HD, TB);
  const int aggBlocks = cdiv((long long)E * 32, TB);

  // --- layer 1 ---
  k_gemm_wmma<FEAT><<<gemmBlocks, TB, 0, stream>>>(x, W1, bufA, N);
  k_fill<<<nhBlocks, TB, 0, stream>>>(bufB, 0.0f, N * HD);
  k_agg<<<aggBlocks, TB, 0, stream>>>(bufA, src, dst, norm, bufB, E);
  k_bias_relu<<<nhBlocks, TB, 0, stream>>>(bufB, bufA, dinv, b1, N);   // h1 in bufB

  // --- layer 2 ---
  k_gemm_wmma<HD><<<gemmBlocks, TB, 0, stream>>>(bufB, W2, bufA, N);
  k_fill<<<nhBlocks, TB, 0, stream>>>(bufB, 0.0f, N * HD);
  k_agg<<<aggBlocks, TB, 0, stream>>>(bufA, src, dst, norm, bufB, E);
  k_bias_relu<<<nhBlocks, TB, 0, stream>>>(bufB, bufA, dinv, b2, N);   // h2 in bufB

  // --- heads ---
  k_logits<<<cdiv(N, TB), TB, 0, stream>>>(bufB, Wa, ba, outv, N);
  k_value<<<NGRAPH, TB, 0, stream>>>(bufB, batch, Wc, bc, outv + N, N);
}
